// GaussianFeatureLeverage_60155311948473
// MI455X (gfx1250) — matrix operand order repaired and checked by reading, not compiled, not run on hardware
//
#include <hip/hip_runtime.h>
#include <hip/hip_bf16.h>

typedef __attribute__((ext_vector_type(16))) _Float16 v16h;
typedef __attribute__((ext_vector_type(8)))  float    v8f;

#define ACT_NONE    0
#define ACT_GELU    1
#define ACT_ELU     2
#define ACT_SIGMOID 3
#define ACT_ABS     4

__device__ __forceinline__ float applyAct(float v, int act) {
  switch (act) {
    case ACT_GELU:    return 0.5f * v * (1.0f + erff(v * 0.70710678118654752440f));
    case ACT_ELU:     return v > 0.0f ? v : expm1f(v);
    case ACT_SIGMOID: return 1.0f / (1.0f + expf(-v));
    case ACT_ABS:     return fabsf(v);
    default:          return v;
  }
}

// ---------------------------------------------------------------------------
// Pre-swizzle conv weights (Cout, Cin*9) f32 row-major into the per-lane
// WMMA A-matrix (16x32 f16) layout:
//   wt[((mt*Ksteps + ks)*32 + lane)*16 + i]
//   lane: M = lane%16, half = lane/16
//   K = ks*32 + 16*half + i
// Zero-pads Cout to 16 and K to 32 (the zero rows make out-of-range K
// contributions vanish, so the conv gather needs no K guard).
// ---------------------------------------------------------------------------
__global__ void swizzleWeightsKernel(const float* __restrict__ w,
                                     _Float16* __restrict__ wt,
                                     int Cout, int Ktot, int Ksteps, int total) {
  int idx = blockIdx.x * blockDim.x + threadIdx.x;
  if (idx >= total) return;
  int i    = idx & 15;
  int lane = (idx >> 4) & 31;
  int t    = idx >> 9;          // mt*Ksteps + ks
  int ks   = t % Ksteps;
  int mt   = t / Ksteps;
  int m    = mt * 16 + (lane & 15);
  int half = lane >> 4;
  int koff = 16 * half + i;
  int k    = ks * 32 + koff;
  float v = 0.0f;
  if (m < Cout && k < Ktot) v = w[(size_t)m * Ktot + k];
  wt[idx] = (_Float16)v;
}

// ---------------------------------------------------------------------------
// Implicit-GEMM 3x3 same-padded conv via V_WMMA_F32_16X16X32_F16.
// One wave computes MT 16(Cout)x16(pixel) tiles sharing a single B gather.
// Branchless hot loop: invalid pixels / K rows load clamped (valid) addresses;
// their contributions are zeroed by the zero-padded A tile and guarded stores.
// PADR: 0 = edge clamp, 1 = numpy-'reflect'.  NHWC: input layout flag.
// ---------------------------------------------------------------------------
template <int PADR, int NHWC, int MT>
__global__ void conv3x3WmmaActKernel(const float* __restrict__ x,
                                     const _Float16* __restrict__ wt,
                                     const float* __restrict__ bias,
                                     float* __restrict__ y,
                                     int Bn, int Cin, int Cout, int Hh, int Ww,
                                     int mTiles, int act) {
  const int lane = threadIdx.x;
  const int half = lane >> 4;
  const int hw   = Hh * Ww;
  const int Npix = Bn * hw;
  const int n    = blockIdx.x * 16 + (lane & 15);
  const bool nValid = (n < Npix);
  const int nc  = nValid ? n : (Npix - 1);
  int bN  = nc / hw;
  int rem = nc - bN * hw;
  int oy  = rem / Ww;
  int ox  = rem - oy * Ww;

  // boundary-resolved neighbor coordinates (branchless selects)
  int oym1, oyp1, oxm1, oxp1;
  if (PADR) {
    oym1 = (oy == 0)      ? 1      : oy - 1;
    oyp1 = (oy == Hh - 1) ? Hh - 2 : oy + 1;
    oxm1 = (ox == 0)      ? 1      : ox - 1;
    oxp1 = (ox == Ww - 1) ? Ww - 2 : ox + 1;
  } else {
    oym1 = (oy == 0)      ? 0      : oy - 1;
    oyp1 = (oy == Hh - 1) ? Hh - 1 : oy + 1;
    oxm1 = (ox == 0)      ? 0      : ox - 1;
    oxp1 = (ox == Ww - 1) ? Ww - 1 : ox + 1;
  }

  const int Ktot   = Cin * 9;
  const int Ksteps = (Ktot + 31) >> 5;
  const int CinM1  = Cin - 1;

  v8f acc[MT];
  const _Float16* ap[MT];
#pragma unroll
  for (int t = 0; t < MT; ++t) {
    acc[t] = (v8f){0.f, 0.f, 0.f, 0.f, 0.f, 0.f, 0.f, 0.f};
    int mt = blockIdx.y * MT + t;
    if (mt > mTiles - 1) mt = mTiles - 1;          // clamp: duplicate A, store guarded
    ap[t] = wt + (size_t)mt * Ksteps * 512 + lane * 16;
  }

  const float* xbase = x + (size_t)bN * hw * Cin;  // same base for NCHW and NHWC
  const int k0base = 16 * half;

  for (int ks = 0; ks < Ksteps; ++ks) {
    v16h aM[MT];
#pragma unroll
    for (int t = 0; t < MT; ++t) {
      aM[t] = *(const v16h*)ap[t];
      ap[t] += 512;
    }
    __builtin_prefetch(ap[0], 0, 1);               // global_prefetch_b8 next A tile

    const int k0 = ks * 32 + k0base;
    float tv[16];
#pragma unroll
    for (int i = 0; i < 16; ++i) {                 // branchless gather, loads clause together
      int k  = k0 + i;
      int ci = k / 9;                              // const-divisor magic
      int r  = k - ci * 9;
      ci = (ci > CinM1) ? CinM1 : ci;              // clamp for K padding (A rows are zero)
      int ky = (r >= 3) + (r >= 6);
      int kx = r - ky * 3;
      int iyv = (ky == 0) ? oym1 : ((ky == 2) ? oyp1 : oy);
      int ixv = (kx == 0) ? oxm1 : ((kx == 2) ? oxp1 : ox);
      int pix = iyv * Ww + ixv;
      tv[i] = NHWC ? xbase[(size_t)pix * Cin + ci]
                   : xbase[(size_t)ci * hw + pix];
    }
    v16h bM;
#pragma unroll
    for (int i = 0; i < 16; ++i) bM[i] = (_Float16)tv[i];

#pragma unroll
    for (int t = 0; t < MT; ++t) {
      acc[t] = __builtin_amdgcn_wmma_f32_16x16x32_f16(
          /*neg_a=*/false, aM[t], /*neg_b=*/false, bM,
          /*c_mod=*/(short)0, acc[t], /*reuse_a=*/false, /*reuse_b=*/false);
    }
  }

  if (!nValid) return;
#pragma unroll
  for (int t = 0; t < MT; ++t) {
    int mt = blockIdx.y * MT + t;
    if (mt >= mTiles) break;
#pragma unroll
    for (int rr = 0; rr < 8; ++rr) {
      int co = mt * 16 + 8 * half + rr;            // C/D layout: M = rr + 8*(lane/16)
      if (co < Cout) {
        float v = acc[t][rr] + bias[co];
        y[(((size_t)bN * Cout + co) * Hh + oy) * Ww + ox] = applyAct(v, act);
      }
    }
  }
}

// ---------------------------------------------------------------------------
// Fused backproject + EWA gaussian rasterize scatter. One thread per gaussian.
// acc: [B*H2*W2, 64] NHWC accumulator, den: [B*H2*W2] weights.
// ---------------------------------------------------------------------------
__global__ void rasterizeKernel(const float* __restrict__ disp,
                                const float* __restrict__ invKall,
                                const float* __restrict__ Kall,
                                const float* __restrict__ rot,
                                const float* __restrict__ scl,
                                const float* __restrict__ opa,
                                const float* __restrict__ gf,
                                float* __restrict__ acc,
                                float* __restrict__ den,
                                int scale, int Bn, int h, int w, int H2, int W2) {
  int g  = blockIdx.x * blockDim.x + threadIdx.x;
  int hw = h * w;
  if (g >= Bn * hw) return;
  int bN  = g / hw;
  int rem = g - bN * hw;
  int gy  = rem / w;
  int gx  = rem - gy * w;

  float depth = 1.0f / (0.01f + 9.99f * disp[g]);
  depth = fminf(fmaxf(depth, 0.1f), 100.0f);

  const float* iK = invKall + ((size_t)scale * Bn + bN) * 16;
  float xf = (float)gx, yf = (float)gy;
  float px = (iK[0] * xf + iK[1] * yf + iK[2])  * depth;
  float py = (iK[4] * xf + iK[5] * yf + iK[6])  * depth;
  float pz = (iK[8] * xf + iK[9] * yf + iK[10]) * depth;

  const float* Km = Kall + ((size_t)(scale + 1) * Bn + bN) * 16;
  float fx = Km[0], fy = Km[5], cx = Km[2], cy = Km[6];
  float z  = fmaxf(pz, 1e-6f);
  float iz = 1.0f / z;
  float u  = fx * px * iz + cx;
  float v  = fy * py * iz + cy;

  float qw = rot[((size_t)bN * 4 + 0) * hw + rem];
  float qx = rot[((size_t)bN * 4 + 1) * hw + rem];
  float qy = rot[((size_t)bN * 4 + 2) * hw + rem];
  float qz = rot[((size_t)bN * 4 + 3) * hw + rem];
  float qn = fmaxf(sqrtf(qw * qw + qx * qx + qy * qy + qz * qz), 1e-12f);
  qw /= qn; qx /= qn; qy /= qn; qz /= qn;

  float s0 = scl[((size_t)bN * 3 + 0) * hw + rem];
  float s1 = scl[((size_t)bN * 3 + 1) * hw + rem];
  float s2 = scl[((size_t)bN * 3 + 2) * hw + rem];

  float R00 = 1.f - 2.f * (qy * qy + qz * qz), R01 = 2.f * (qx * qy - qw * qz), R02 = 2.f * (qx * qz + qw * qy);
  float R10 = 2.f * (qx * qy + qw * qz), R11 = 1.f - 2.f * (qx * qx + qz * qz), R12 = 2.f * (qy * qz - qw * qx);
  float R20 = 2.f * (qx * qz - qw * qy), R21 = 2.f * (qy * qz + qw * qx), R22 = 1.f - 2.f * (qx * qx + qy * qy);
  float M00 = R00 * s0, M01 = R01 * s1, M02 = R02 * s2;
  float M10 = R10 * s0, M11 = R11 * s1, M12 = R12 * s2;
  float M20 = R20 * s0, M21 = R21 * s1, M22 = R22 * s2;
  float C00 = M00 * M00 + M01 * M01 + M02 * M02;
  float C01 = M00 * M10 + M01 * M11 + M02 * M12;
  float C02 = M00 * M20 + M01 * M21 + M02 * M22;
  float C11 = M10 * M10 + M11 * M11 + M12 * M12;
  float C12 = M10 * M20 + M11 * M21 + M12 * M22;
  float C22 = M20 * M20 + M21 * M21 + M22 * M22;

  float J00 = fx * iz, J02 = -fx * px * iz * iz;
  float J11 = fy * iz, J12 = -fy * py * iz * iz;
  float A00 = J00 * C00 + J02 * C02, A01 = J00 * C01 + J02 * C12, A02 = J00 * C02 + J02 * C22;
  float A10 = J11 * C01 + J12 * C02, A11 = J11 * C11 + J12 * C12, A12 = J11 * C12 + J12 * C22;
  float c2a = A00 * J00 + A02 * J02 + 0.3f;
  float c2b = A01 * J11 + A02 * J12;
  float c2c = A11 * J11 + A12 * J12 + 0.3f;
  float det = fmaxf(c2a * c2c - c2b * c2b, 1e-6f);
  float ia = c2c / det, ib = -c2b / det, ic = c2a / det;

  float o = opa[(size_t)bN * hw + rem];

  float f[64];
#pragma unroll
  for (int c = 0; c < 64; ++c) f[c] = gf[((size_t)bN * 64 + c) * hw + rem];

  float ur = rintf(u), vr = rintf(v);  // jnp.round = ties-to-even
  for (int dy = -1; dy <= 1; ++dy) {
    for (int dx = -1; dx <= 1; ++dx) {
      float uc = ur + (float)dx, vc = vr + (float)dy;
      if (uc < 0.f || uc >= (float)W2 || vc < 0.f || vc >= (float)H2) continue;
      float du = uc - u, dv = vc - v;
      float power = -0.5f * (ia * du * du + 2.f * ib * du * dv + ic * dv * dv);
      float wgt = o * expf(fminf(power, 0.f));
      int ui = (int)uc, vi = (int)vc;
      size_t idx = (size_t)bN * H2 * W2 + (size_t)vi * W2 + ui;
      __hip_atomic_fetch_add(&den[idx], wgt, __ATOMIC_RELAXED, __HIP_MEMORY_SCOPE_AGENT);
      float* accp = acc + idx * 64;
#pragma unroll
      for (int c = 0; c < 64; ++c)
        __hip_atomic_fetch_add(&accp[c], wgt * f[c], __ATOMIC_RELAXED, __HIP_MEMORY_SCOPE_AGENT);
    }
  }
}

__global__ void normalizeAccKernel(float* __restrict__ acc,
                                   const float* __restrict__ den, int total) {
  int i = blockIdx.x * blockDim.x + threadIdx.x;
  if (i >= total) return;
  acc[i] = acc[i] / (den[i >> 6] + 1e-8f);
}

// ---------------------------------------------------------------------------
// Host orchestration
// ---------------------------------------------------------------------------
static inline void launchConv(hipStream_t stream, const float* x, const float* w,
                              const float* bias, float* y, _Float16* wswz,
                              int Bn, int Cin, int Cout, int Hh, int Ww,
                              int padReflect, int act, int inNHWC) {
  int Ktot   = Cin * 9;
  int Ksteps = (Ktot + 31) / 32;
  int mTiles = (Cout + 15) / 16;
  int total  = mTiles * Ksteps * 512;
  swizzleWeightsKernel<<<(total + 255) / 256, 256, 0, stream>>>(w, wswz, Cout, Ktot, Ksteps, total);

  int MT = (mTiles >= 2) ? 4 : 1;
  dim3 grid((Bn * Hh * Ww + 15) / 16, (mTiles + MT - 1) / MT);
  if (MT == 1) {
    // mTiles==1 launches are the small head convs: always edge-pad NCHW
    conv3x3WmmaActKernel<0, 0, 1><<<grid, 32, 0, stream>>>(x, wswz, bias, y, Bn, Cin, Cout,
                                                           Hh, Ww, mTiles, act);
  } else if (inNHWC) {
    conv3x3WmmaActKernel<1, 1, 4><<<grid, 32, 0, stream>>>(x, wswz, bias, y, Bn, Cin, Cout,
                                                           Hh, Ww, mTiles, act);
  } else {
    conv3x3WmmaActKernel<1, 0, 4><<<grid, 32, 0, stream>>>(x, wswz, bias, y, Bn, Cin, Cout,
                                                           Hh, Ww, mTiles, act);
  }
}

extern "C" void kernel_launch(void* const* d_in, const int* in_sizes, int n_in,
                              void* d_out, int out_size, void* d_ws, size_t ws_size,
                              hipStream_t stream) {
  (void)in_sizes; (void)n_in; (void)out_size; (void)ws_size;
  const int Bn = 8;
  const int CH_IN[4]  = {64, 128, 256, 512};
  const int CH_OUT[5] = {64, 18, 36, 72, 144};
  const int HS[4] = {48, 24, 12, 6};
  const int WS[4] = {160, 80, 40, 20};

  auto F = [&](int i) { return (const float*)d_in[i]; };
  // params flattened JAX-style (sorted keys, depth-first). Per scale i (base = i*18):
  //  +0 lev1.b  +1 lev1.w  +2 lev2.b  +3 lev2.w
  //  +4 opa.b1  +5 opa.b2  +6 opa.w1  +7 opa.w2
  //  +8 res.b   +9 res.w   +10 rot.b1 +11 rot.b2 +12 rot.w1 +13 rot.w2
  //  +14 scl.b1 +15 scl.b2 +16 scl.w1 +17 scl.w2
  // 72: s4.res.b  73: s4.res.w  74..81: feat0,disp0,..,feat3,disp3  82: inv_K  83: K
  const int IDX_FEAT0 = 74;
  const float* invK = F(82);
  const float* Kmat = F(83);

  uint8_t* ws = (uint8_t*)d_ws;
  size_t off = 0;
  auto wsAlloc = [&](size_t bytes) -> void* {
    void* p = ws + off;
    off += (bytes + 255) & ~(size_t)255;
    return p;
  };
  _Float16* wswz = (_Float16*)wsAlloc((size_t)2359296 * sizeof(_Float16)); // 512x4608 max
  float* bufH1   = (float*)wsAlloc((size_t)8 * 8 * 48 * 160 * sizeof(float));
  float* bufRot  = (float*)wsAlloc((size_t)8 * 4 * 48 * 160 * sizeof(float));
  float* bufScl  = (float*)wsAlloc((size_t)8 * 3 * 48 * 160 * sizeof(float));
  float* bufOpa  = (float*)wsAlloc((size_t)8 * 1 * 48 * 160 * sizeof(float));
  float* bufLev1 = (float*)wsAlloc((size_t)8 * 64 * 48 * 160 * sizeof(float));
  float* bufGf   = (float*)wsAlloc((size_t)8 * 64 * 48 * 160 * sizeof(float));
  float* bufAcc  = (float*)wsAlloc((size_t)8 * 96 * 320 * 64 * sizeof(float));
  float* bufDen  = (float*)wsAlloc((size_t)8 * 96 * 320 * sizeof(float));

  size_t outOff[5];
  outOff[0] = 0;
  outOff[1] = outOff[0] + (size_t)8 * 64 * 96 * 320;
  outOff[2] = outOff[1] + (size_t)8 * 18 * 48 * 160;
  outOff[3] = outOff[2] + (size_t)8 * 36 * 24 * 80;
  outOff[4] = outOff[3] + (size_t)8 * 72 * 12 * 40;
  float* out = (float*)d_out;

  for (int i = 0; i < 4; ++i) {
    int base = i * 18;
    const float* feat = F(IDX_FEAT0 + 2 * i);
    const float* disp = F(IDX_FEAT0 + 2 * i + 1);
    int h = HS[i], w = WS[i], c = CH_IN[i];
    int H2 = 2 * h, W2 = 2 * w;

    // rot head: conv -> GELU -> conv
    launchConv(stream, feat,  F(base + 12), F(base + 10), bufH1,  wswz, Bn, c, 8, h, w, 0, ACT_GELU, 0);
    launchConv(stream, bufH1, F(base + 13), F(base + 11), bufRot, wswz, Bn, 8, 4, h, w, 0, ACT_NONE, 0);
    // scl head: conv -> GELU -> conv -> abs
    launchConv(stream, feat,  F(base + 16), F(base + 14), bufH1,  wswz, Bn, c, 6, h, w, 0, ACT_GELU, 0);
    launchConv(stream, bufH1, F(base + 17), F(base + 15), bufScl, wswz, Bn, 6, 3, h, w, 0, ACT_ABS, 0);
    // opa head: conv -> GELU -> conv -> sigmoid
    launchConv(stream, feat,  F(base + 6),  F(base + 4),  bufH1,  wswz, Bn, c, 2, h, w, 0, ACT_GELU, 0);
    launchConv(stream, bufH1, F(base + 7),  F(base + 5),  bufOpa, wswz, Bn, 2, 1, h, w, 0, ACT_SIGMOID, 0);
    // feature ConvBlocks (reflect pad + ELU)
    launchConv(stream, feat,    F(base + 1), F(base + 0), bufLev1, wswz, Bn, c, c,  h, w, 1, ACT_ELU, 0);
    launchConv(stream, bufLev1, F(base + 3), F(base + 2), bufGf,   wswz, Bn, c, 64, h, w, 1, ACT_ELU, 0);

    // rasterize at 2x resolution
    size_t npix2 = (size_t)Bn * H2 * W2;
    (void)hipMemsetAsync(bufAcc, 0, npix2 * 64 * sizeof(float), stream);
    (void)hipMemsetAsync(bufDen, 0, npix2 * sizeof(float), stream);
    int Ngs = Bn * h * w;
    rasterizeKernel<<<(Ngs + 255) / 256, 256, 0, stream>>>(
        disp, invK, Kmat, bufRot, bufScl, bufOpa, bufGf, bufAcc, bufDen,
        i, Bn, h, w, H2, W2);
    int totN = (int)(npix2 * 64);
    normalizeAccKernel<<<(totN + 255) / 256, 256, 0, stream>>>(bufAcc, bufDen, totN);

    // residual ConvBlock reading the NHWC accumulator, writing straight to d_out
    launchConv(stream, bufAcc, F(base + 9), F(base + 8), out + outOff[i], wswz,
               Bn, 64, CH_OUT[i], H2, W2, 1, ACT_ELU, 1);
  }

  // s4: ConvBlock on feat3 (512 -> 144)
  launchConv(stream, F(IDX_FEAT0 + 6), F(73), F(72), out + outOff[4], wswz,
             Bn, 512, 144, HS[3], WS[3], 1, ACT_ELU, 0);
}
